// DynamicWSIGraphEncoder_20066087206937
// MI455X (gfx1250) — matrix operand order repaired and checked by reading, not compiled, not run on hardware
//
#include <hip/hip_runtime.h>

// ---------------- problem constants (from reference setup_inputs) ------------
#define B_ 4
#define N_ 4096
#define D_ 1024
#define H_ 512
#define A_ 256
#define K_ 8
#define BIGF 1e9f

typedef __attribute__((ext_vector_type(16))) __bf16 v16bf;
typedef __attribute__((ext_vector_type(8)))  __bf16 v8bf;
typedef __attribute__((ext_vector_type(8)))  float  v8f;

#define WPB 4   // waves per block for WMMA kernels (128 threads)

// ---------------- WMMA tile helpers (layouts per CDNA5 ISA) -------
// A matrix 16x32 bf16, row-major source: lane = 32 lanes; row = lane&15.
// lanes 0-15 hold K = {0..7, 16..23}; lanes 16-31 hold K = {8..15, 24..31}.
__device__ __forceinline__ v16bf load_tile_a(const __bf16* base, int row,
                                             int ldk, int kbase, int lane) {
  const __bf16* p = base + (size_t)row * ldk + kbase + ((lane & 16) ? 8 : 0);
  v8bf lo = *(const v8bf*)p;
  v8bf hi = *(const v8bf*)(p + 16);
  v16bf a;
#pragma unroll
  for (int i = 0; i < 8; ++i) { a[i] = lo[i]; a[8 + i] = hi[i]; }
  return a;
}

// B matrix 32x16 bf16 from a K-contiguous (column-major weight) source:
// lane's column = col (lane&15); lanes 0-15 hold K={0..7,16..23}, lanes 16-31 K={8..15,24..31}.
__device__ __forceinline__ v16bf load_tile_b(const __bf16* base, int col,
                                             int ldk, int kbase, int lane) {
  const __bf16* p = base + (size_t)col * ldk + kbase + ((lane & 16) ? 8 : 0);
  v8bf lo = *(const v8bf*)p;
  v8bf hi = *(const v8bf*)(p + 16);
  v16bf b;
#pragma unroll
  for (int i = 0; i < 8; ++i) { b[i] = lo[i]; b[8 + i] = hi[i]; }
  return b;
}

__device__ __forceinline__ v8f wmma_bf16(v16bf a, v16bf b, v8f c) {
  return __builtin_amdgcn_wmma_f32_16x16x32_bf16(
      /*neg_a=*/false, a, /*neg_b=*/false, b,
      /*c_mod=*/(short)0, c, /*reuse_a=*/false, /*reuse_b=*/false);
}

// ---------------- small utility kernels --------------------------------------
__global__ void cvt_bf16_kernel(const float* __restrict__ in,
                                __bf16* __restrict__ out, size_t n) {
  size_t i = (size_t)blockIdx.x * blockDim.x + threadIdx.x;
  if (i < n) out[i] = (__bf16)in[i];
}

// W [kdim][cdim] fp32 row-major  ->  Wt [cdim][kdim] bf16 (K-contiguous)
__global__ void transpose_cvt_kernel(const float* __restrict__ w,
                                     __bf16* __restrict__ wt,
                                     int kdim, int cdim) {
  size_t idx = (size_t)blockIdx.x * blockDim.x + threadIdx.x;
  if (idx >= (size_t)kdim * cdim) return;
  int c = (int)(idx / kdim);
  int k = (int)(idx % kdim);
  wt[idx] = (__bf16)w[(size_t)k * cdim + c];
}

// squared row norms of bf16 feature rows (matches the bf16 Gram path)
__global__ void sqnorm_kernel(const __bf16* __restrict__ feat,
                              float* __restrict__ sq, int dim) {
  int row = blockIdx.x;
  const __bf16* f = feat + (size_t)row * dim;
  float acc = 0.f;
  for (int d = threadIdx.x; d < dim; d += blockDim.x) {
    float v = (float)f[d];
    acc += v * v;
  }
  __shared__ float red[256];
  red[threadIdx.x] = acc;
  __syncthreads();
  for (int off = 128; off > 0; off >>= 1) {
    if (threadIdx.x < off) red[threadIdx.x] += red[threadIdx.x + off];
    __syncthreads();
  }
  if (threadIdx.x == 0) sq[row] = red[0];
}

// ---------------- streaming kNN with WMMA Gram tiles + register top-8 --------
// One wave owns 16 rows of one batch; all 4 waves of a block stream the SAME
// column tiles, which are staged cooperatively into LDS with async global->LDS
// copies (ASYNCcnt path) and consumed as WMMA B-operands via ds_load_b128.
// The k-loop is software-pipelined (double-buffered A/B register tiles).
__global__ void knn_kernel(const __bf16* __restrict__ feat,
                           const float* __restrict__ sq,
                           const float* __restrict__ mask,
                           int* __restrict__ nbr, int dim) {
  __shared__ __align__(16) __bf16 colbuf[16 * D_];   // 32 KB column-tile stage
  __shared__ float ldsd[WPB][16][17];
  __shared__ int   ldsi[WPB][16][8];

  const int wave = threadIdx.x >> 5;
  const int lane = threadIdx.x & 31;
  const int gw = blockIdx.x * WPB + wave;          // 0 .. B*N/16-1
  const int tilesPerBatch = N_ / 16;
  const int b = gw / tilesPerBatch;
  const int rowBase = (gw % tilesPerBatch) * 16;   // batch-local; same batch for
                                                   // all 4 waves of a block
  const __bf16* fb = feat + (size_t)b * N_ * dim;
  const float*  sqb = sq + b * N_;
  const float*  mb  = mask + b * N_;

  const int r = lane & 15;          // row this lane owns for top-k
  const int colhalf = lane >> 4;    // which 8 columns of each tile this lane scans

  const unsigned cbase = (unsigned)(uintptr_t)&colbuf[0];  // LDS byte offset
  const int cpIters = dim >> 6;     // (16*dim*2 bytes) / (128 lanes * 16B)

  float sqr8[8];
#pragma unroll
  for (int i = 0; i < 8; ++i)
    sqr8[i] = sqb[rowBase + i + ((lane & 16) ? 8 : 0)];

  float bestd[8];
  int   besti[8];
#pragma unroll
  for (int j = 0; j < 8; ++j) { bestd[j] = 3e38f; besti[j] = 0; }

  for (int ct = 0; ct < N_; ct += 16) {
    // ---- stage contiguous 16 x dim bf16 column tile into LDS (async) ----
    const __bf16* srcbase = fb + (size_t)ct * dim;   // 16 consecutive node rows
    for (int it = 0; it < cpIters; ++it) {
      unsigned byteoff = (unsigned)((it * (WPB * 32) + threadIdx.x) * 16);
      unsigned ldsoff = cbase + byteoff;
      asm volatile("global_load_async_to_lds_b128 %0, %1, %2"
                   :: "v"(ldsoff), "v"(byteoff), "s"(srcbase)
                   : "memory");
    }
    asm volatile("s_wait_asynccnt 0x0" ::: "memory");
    __syncthreads();  // colbuf ready for all waves

    // ---- double-buffered WMMA k-loop ----
    v8f c = {};
    v16bf a0 = load_tile_a(fb, rowBase + (lane & 15), dim, 0, lane);
    v16bf b0 = load_tile_b(colbuf, lane & 15, dim, 0, lane);
    for (int k = 32; k < dim; k += 32) {
      v16bf a1 = load_tile_a(fb, rowBase + (lane & 15), dim, k, lane);
      v16bf b1 = load_tile_b(colbuf, lane & 15, dim, k, lane);
      c = wmma_bf16(a0, b0, c);
      a0 = a1; b0 = b1;
    }
    c = wmma_bf16(a0, b0, c);

    const int ccol = ct + (lane & 15);
    const float sqc = sqb[ccol];
    const float mpen = (mb[ccol] <= 0.f) ? BIGF : 0.f;
#pragma unroll
    for (int i = 0; i < 8; ++i) {
      const int rr = i + ((lane & 16) ? 8 : 0);
      float d2 = sqr8[i] + sqc - 2.f * c[i] + mpen;
      if (rowBase + rr == ccol) d2 += BIGF;  // exclude self
      ldsd[wave][rr][lane & 15] = d2;
    }
    __syncthreads();  // also guarantees all colbuf reads are done

#pragma unroll
    for (int n = 0; n < 8; ++n) {
      const int cn = colhalf * 8 + n;
      float d = ldsd[wave][r][cn];
      int  ci = ct + cn;
      if (d < bestd[7]) {
#pragma unroll
        for (int j = 0; j < 8; ++j) {
          if (d < bestd[j]) {
            float td = bestd[j]; int ti = besti[j];
            bestd[j] = d; besti[j] = ci; d = td; ci = ti;
          }
        }
      }
    }
    __syncthreads();
  }

  // merge the two column-half heaps for each row
  if (colhalf == 1) {
#pragma unroll
    for (int j = 0; j < 8; ++j) { ldsd[wave][r][j] = bestd[j]; ldsi[wave][r][j] = besti[j]; }
  }
  __syncthreads();
  if (colhalf == 0) {
#pragma unroll
    for (int n = 0; n < 8; ++n) {
      float d = ldsd[wave][r][n];
      int  ci = ldsi[wave][r][n];
      if (d < bestd[7]) {
#pragma unroll
        for (int j = 0; j < 8; ++j) {
          if (d < bestd[j]) {
            float td = bestd[j]; int ti = besti[j];
            bestd[j] = d; besti[j] = ci; d = td; ci = ti;
          }
        }
      }
    }
    int* dst = nbr + ((size_t)b * N_ + rowBase + r) * K_;
#pragma unroll
    for (int j = 0; j < K_; ++j) dst[j] = besti[j];
  }
}

// neigh[b,n,:] = mask(row) * mean_{j<8} feat[b, idx[j], :]   (deg == k == 8)
__global__ void gather_mean_kernel(const __bf16* __restrict__ feat,
                                   const int* __restrict__ nbr,
                                   const float* __restrict__ mask,
                                   __bf16* __restrict__ out, int dim) {
  const int node = blockIdx.x;                 // 0 .. B*N-1
  const int b = node / N_;
  const __bf16* fb = feat + (size_t)b * N_ * dim;
  const int* nb = nbr + (size_t)node * K_;
  int idx[K_];
#pragma unroll
  for (int j = 0; j < K_; ++j) idx[j] = nb[j];
  const float m = (mask[node] > 0.f) ? 0.125f : 0.f;
  for (int d = threadIdx.x; d < dim; d += blockDim.x) {
    float acc = 0.f;
#pragma unroll
    for (int j = 0; j < K_; ++j) acc += (float)fb[(size_t)idx[j] * dim + d];
    out[(size_t)node * dim + d] = (__bf16)(acc * m);
  }
}

// out = relu([x | neigh] @ W + b); one wave computes a 16x64 output tile.
// A-operand is double-buffered across the two K phases (x, then neigh).
__global__ void sage_gemm_kernel(const __bf16* __restrict__ xa,
                                 const __bf16* __restrict__ na,
                                 const __bf16* __restrict__ wt,  // [H][2*dim]
                                 const float* __restrict__ bias,
                                 float* __restrict__ outf,
                                 __bf16* __restrict__ outbf, int dim) {
  const int wave = threadIdx.x >> 5;
  const int lane = threadIdx.x & 31;
  const int gw = blockIdx.x * WPB + wave;
  const int colGroups = H_ / 64;
  const int rowBase = (gw / colGroups) * 16;   // over flattened B*N
  const int colBase = (gw % colGroups) * 64;
  const int ld = 2 * dim;
  const int arow = rowBase + (lane & 15);

  v8f cc[4] = {};
  v16bf a0 = load_tile_a(xa, arow, dim, 0, lane);
  // phase 1: x @ W_top
  for (int k = 0; k < dim - 32; k += 32) {
    v16bf a1 = load_tile_a(xa, arow, dim, k + 32, lane);
#pragma unroll
    for (int t = 0; t < 4; ++t)
      cc[t] = wmma_bf16(a0, load_tile_b(wt, colBase + 16 * t, ld, k, lane), cc[t]);
    a0 = a1;
  }
  {
    v16bf a1 = load_tile_a(na, arow, dim, 0, lane);  // prefetch phase-2 head
#pragma unroll
    for (int t = 0; t < 4; ++t)
      cc[t] = wmma_bf16(a0, load_tile_b(wt, colBase + 16 * t, ld, dim - 32, lane), cc[t]);
    a0 = a1;
  }
  // phase 2: neigh @ W_bot
  for (int k = 0; k < dim - 32; k += 32) {
    v16bf a1 = load_tile_a(na, arow, dim, k + 32, lane);
#pragma unroll
    for (int t = 0; t < 4; ++t)
      cc[t] = wmma_bf16(a0, load_tile_b(wt, colBase + 16 * t, ld, dim + k, lane), cc[t]);
    a0 = a1;
  }
#pragma unroll
  for (int t = 0; t < 4; ++t)
    cc[t] = wmma_bf16(a0, load_tile_b(wt, colBase + 16 * t, ld, 2 * dim - 32, lane), cc[t]);

#pragma unroll
  for (int t = 0; t < 4; ++t) {
    const int col = colBase + 16 * t + (lane & 15);
    const float bv = bias[col];
#pragma unroll
    for (int i = 0; i < 8; ++i) {
      const int row = rowBase + i + ((lane & 16) ? 8 : 0);
      float v = cc[t][i] + bv;
      v = fmaxf(v, 0.f);
      outf[(size_t)row * H_ + col]  = v;
      outbf[(size_t)row * H_ + col] = (__bf16)v;
    }
  }
}

// scores[b,n] = (tanh(feat @ W1 + b1) @ W2 + b2), masked.
__global__ void pool_scores_kernel(const __bf16* __restrict__ feat,
                                   const __bf16* __restrict__ wt,  // [A][dim]
                                   const float* __restrict__ ba,
                                   const float* __restrict__ w2,
                                   const float* __restrict__ b2,
                                   const float* __restrict__ mask,
                                   float* __restrict__ scores, int dim) {
  __shared__ float lds[WPB][32][8];
  const int wave = threadIdx.x >> 5;
  const int lane = threadIdx.x & 31;
  const int gw = blockIdx.x * WPB + wave;    // 0 .. B*N/16-1
  const int rowBase = gw * 16;               // over flattened B*N

  float acc[8] = {0.f, 0.f, 0.f, 0.f, 0.f, 0.f, 0.f, 0.f};
  for (int ctile = 0; ctile < A_; ctile += 16) {
    v8f c = {};
    v16bf a0 = load_tile_a(feat, rowBase + (lane & 15), dim, 0, lane);
    v16bf b0 = load_tile_b(wt, ctile + (lane & 15), dim, 0, lane);
    for (int k = 32; k < dim; k += 32) {
      v16bf a1 = load_tile_a(feat, rowBase + (lane & 15), dim, k, lane);
      v16bf b1 = load_tile_b(wt, ctile + (lane & 15), dim, k, lane);
      c = wmma_bf16(a0, b0, c);
      a0 = a1; b0 = b1;
    }
    c = wmma_bf16(a0, b0, c);

    const int col = ctile + (lane & 15);
    const float bav = ba[col], w2v = w2[col];
#pragma unroll
    for (int i = 0; i < 8; ++i) acc[i] += tanhf(c[i] + bav) * w2v;
  }
#pragma unroll
  for (int i = 0; i < 8; ++i) lds[wave][lane][i] = acc[i];
  __syncthreads();
  if (lane < 16) {
    const int r = lane;
    float s = 0.f;
    for (int n = 0; n < 16; ++n)
      s += (r < 8) ? lds[wave][n][r] : lds[wave][n + 16][r - 8];
    s += b2[0];
    const int row = rowBase + r;
    if (mask[row] <= 0.f) s = -1e9f;
    scores[row] = s;
  }
}

__global__ void softmax_kernel(const float* __restrict__ scores,
                               float* __restrict__ attn) {
  const int b = blockIdx.x;
  const float* s = scores + b * N_;
  __shared__ float red[256];
  float m = -3e38f;
  for (int n = threadIdx.x; n < N_; n += blockDim.x) m = fmaxf(m, s[n]);
  red[threadIdx.x] = m;
  __syncthreads();
  for (int off = 128; off > 0; off >>= 1) {
    if (threadIdx.x < off) red[threadIdx.x] = fmaxf(red[threadIdx.x], red[threadIdx.x + off]);
    __syncthreads();
  }
  m = red[0];
  __syncthreads();
  float sum = 0.f;
  for (int n = threadIdx.x; n < N_; n += blockDim.x) sum += __expf(s[n] - m);
  red[threadIdx.x] = sum;
  __syncthreads();
  for (int off = 128; off > 0; off >>= 1) {
    if (threadIdx.x < off) red[threadIdx.x] += red[threadIdx.x + off];
    __syncthreads();
  }
  sum = red[0];
  const float inv = 1.f / sum;
  for (int n = threadIdx.x; n < N_; n += blockDim.x)
    attn[b * N_ + n] = __expf(s[n] - m) * inv;
}

// out[b,:] = sum_n attn[b,n] * src[b,n,:]   (fp32 source, as in reference)
__global__ void weighted_sum_kernel(const float* __restrict__ src,
                                    const float* __restrict__ attn,
                                    float* __restrict__ dst,
                                    float* __restrict__ dst2, int dim) {
  const int b = blockIdx.y;
  const int d = blockIdx.x * blockDim.x + threadIdx.x;
  const float* sb = src + (size_t)b * N_ * dim;
  const float* ab = attn + b * N_;
  float acc = 0.f;
  for (int n = 0; n < N_; ++n) acc += ab[n] * sb[(size_t)n * dim + d];
  dst[b * dim + d] = acc;
  if (dst2) dst2[b * dim + d] = acc;
}

// ---------------- host-side orchestration ------------------------------------
extern "C" void kernel_launch(void* const* d_in, const int* in_sizes, int n_in,
                              void* d_out, int out_size, void* d_ws, size_t ws_size,
                              hipStream_t stream) {
  (void)in_sizes; (void)n_in; (void)out_size; (void)ws_size;

  const float* x    = (const float*)d_in[0];
  const float* mask = (const float*)d_in[1];
  // d_in[2] = coords (unused: dynamic_graph -> kNN from features)
  const float* Wpi1 = (const float*)d_in[3];
  const float* bpi1 = (const float*)d_in[4];
  const float* Wpi2 = (const float*)d_in[5];
  const float* bpi2 = (const float*)d_in[6];
  const float* Wl0  = (const float*)d_in[7];
  const float* bl0  = (const float*)d_in[8];
  const float* Wl1  = (const float*)d_in[9];
  const float* bl1  = (const float*)d_in[10];
  const float* Wq0a = (const float*)d_in[11];
  const float* bq0a = (const float*)d_in[12];
  const float* Wq0b = (const float*)d_in[13];
  const float* bq0b = (const float*)d_in[14];
  const float* Wq1a = (const float*)d_in[15];
  const float* bq1a = (const float*)d_in[16];
  const float* Wq1b = (const float*)d_in[17];
  const float* bq1b = (const float*)d_in[18];

  // output regions (flat fp32, tuple order)
  float* out     = (float*)d_out;
  float* out_raw = out;                                   // [B, D]
  float* out_n0  = out_raw + (size_t)B_ * D_;             // [B, N, H]
  float* out_n1  = out_n0 + (size_t)B_ * N_ * H_;         // [B, N, H]
  float* out_p0  = out_n1 + (size_t)B_ * N_ * H_;         // [B, H]
  float* out_p1  = out_p0 + (size_t)B_ * H_;              // [B, H]
  float* out_fin = out_p1 + (size_t)B_ * H_;              // [B, H]

  // workspace carve-out
  char* wsp = (char*)d_ws;
  size_t off = 0;
  auto take = [&](size_t bytes) -> void* {
    void* p = wsp + off;
    off += (bytes + 255) & ~(size_t)255;
    return p;
  };
  __bf16* xbf    = (__bf16*)take((size_t)B_ * N_ * D_ * 2);
  __bf16* h1bf   = (__bf16*)take((size_t)B_ * N_ * H_ * 2);
  __bf16* h2bf   = (__bf16*)take((size_t)B_ * N_ * H_ * 2);
  __bf16* neigh  = (__bf16*)take((size_t)B_ * N_ * D_ * 2);  // layer1 reuses prefix
  int*    nbr    = (int*)take((size_t)B_ * N_ * K_ * 4);
  float*  sq     = (float*)take((size_t)B_ * N_ * 4);
  float*  scores = (float*)take((size_t)B_ * N_ * 4);
  float*  attn   = (float*)take((size_t)B_ * N_ * 4);
  __bf16* WtL0   = (__bf16*)take((size_t)H_ * 2 * D_ * 2);   // [H][2D]
  __bf16* WtL1   = (__bf16*)take((size_t)H_ * 2 * H_ * 2);   // [H][2H]
  __bf16* WtPI   = (__bf16*)take((size_t)A_ * D_ * 2);       // [A][D]
  __bf16* WtQ0   = (__bf16*)take((size_t)A_ * H_ * 2);       // [A][H]
  __bf16* WtQ1   = (__bf16*)take((size_t)A_ * H_ * 2);       // [A][H]

  const int BLK = 256;
  const int WMMA_BLK = WPB * 32;                 // 128
  const int waveTiles = B_ * N_ / 16;            // 1024 row tiles
  const int knnBlocks  = waveTiles / WPB;        // 256
  const int sageBlocks = waveTiles * (H_ / 64) / WPB;  // 2048
  const int poolBlocks = waveTiles / WPB;        // 256

  // 0) precision conversions
  {
    size_t n = (size_t)B_ * N_ * D_;
    cvt_bf16_kernel<<<(n + BLK - 1) / BLK, BLK, 0, stream>>>(x, xbf, n);
  }
  transpose_cvt_kernel<<<((size_t)2 * D_ * H_ + BLK - 1) / BLK, BLK, 0, stream>>>(Wl0, WtL0, 2 * D_, H_);
  transpose_cvt_kernel<<<((size_t)2 * H_ * H_ + BLK - 1) / BLK, BLK, 0, stream>>>(Wl1, WtL1, 2 * H_, H_);
  transpose_cvt_kernel<<<((size_t)D_ * A_ + BLK - 1) / BLK, BLK, 0, stream>>>(Wpi1, WtPI, D_, A_);
  transpose_cvt_kernel<<<((size_t)H_ * A_ + BLK - 1) / BLK, BLK, 0, stream>>>(Wq0a, WtQ0, H_, A_);
  transpose_cvt_kernel<<<((size_t)H_ * A_ + BLK - 1) / BLK, BLK, 0, stream>>>(Wq1a, WtQ1, H_, A_);

  // 1) raw attention pool on input features
  pool_scores_kernel<<<poolBlocks, WMMA_BLK, 0, stream>>>(xbf, WtPI, bpi1, Wpi2, bpi2, mask, scores, D_);
  softmax_kernel<<<B_, BLK, 0, stream>>>(scores, attn);
  weighted_sum_kernel<<<dim3(D_ / BLK, B_), BLK, 0, stream>>>(x, attn, out_raw, nullptr, D_);

  // 2) layer 0: dynamic kNN on x -> SAGE
  sqnorm_kernel<<<B_ * N_, BLK, 0, stream>>>(xbf, sq, D_);
  knn_kernel<<<knnBlocks, WMMA_BLK, 0, stream>>>(xbf, sq, mask, nbr, D_);
  gather_mean_kernel<<<B_ * N_, BLK, 0, stream>>>(xbf, nbr, mask, neigh, D_);
  sage_gemm_kernel<<<sageBlocks, WMMA_BLK, 0, stream>>>(xbf, neigh, WtL0, bl0, out_n0, h1bf, D_);

  // 3) pool 0
  pool_scores_kernel<<<poolBlocks, WMMA_BLK, 0, stream>>>(h1bf, WtQ0, bq0a, Wq0b, bq0b, mask, scores, H_);
  softmax_kernel<<<B_, BLK, 0, stream>>>(scores, attn);
  weighted_sum_kernel<<<dim3(H_ / BLK, B_), BLK, 0, stream>>>(out_n0, attn, out_p0, nullptr, H_);

  // 4) layer 1: dynamic kNN on h1 -> SAGE
  sqnorm_kernel<<<B_ * N_, BLK, 0, stream>>>(h1bf, sq, H_);
  knn_kernel<<<knnBlocks, WMMA_BLK, 0, stream>>>(h1bf, sq, mask, nbr, H_);
  gather_mean_kernel<<<B_ * N_, BLK, 0, stream>>>(h1bf, nbr, mask, neigh, H_);
  sage_gemm_kernel<<<sageBlocks, WMMA_BLK, 0, stream>>>(h1bf, neigh, WtL1, bl1, out_n1, h2bf, H_);

  // 5) pool 1 (+ final_pooled duplicate)
  pool_scores_kernel<<<poolBlocks, WMMA_BLK, 0, stream>>>(h2bf, WtQ1, bq1a, Wq1b, bq1b, mask, scores, H_);
  softmax_kernel<<<B_, BLK, 0, stream>>>(scores, attn);
  weighted_sum_kernel<<<dim3(H_ / BLK, B_), BLK, 0, stream>>>(out_n1, attn, out_p1, out_fin, H_);
}
